// HGNN_6227702579536
// MI455X (gfx1250) — compile-verified, matrix operand order checked by reading, not run on hardware
//
#include <hip/hip_runtime.h>
#include <hip/hip_bf16.h>

typedef __bf16 v16bf  __attribute__((ext_vector_type(16)));
typedef __bf16 bf16x8 __attribute__((ext_vector_type(8)));
typedef __bf16 bf16x2 __attribute__((ext_vector_type(2)));
typedef float  v8f    __attribute__((ext_vector_type(8)));

#define N_NODES  50000
#define N_HEDGES 50000
#define NNZV     800000
#define IN_CH    512
#define HID_CH   256
#define OUT_CH   64

#define A_RS 40              // padded LDS row stride (halves) for A tile (32 + 8)
#define W_PAD 8              // padding halves per W row

// ---------------- utility kernels ----------------

__global__ __launch_bounds__(256) void k_zero(float* __restrict__ p, long long n) {
    long long i = (long long)blockIdx.x * 256 + threadIdx.x;
    if (i < n) p[i] = 0.0f;
}

__global__ __launch_bounds__(256) void k_degrees(const int* __restrict__ ni,
                                                 const int* __restrict__ hi,
                                                 float* __restrict__ D,
                                                 float* __restrict__ B, int nnz) {
    int i = blockIdx.x * 256 + threadIdx.x;
    if (i < nnz) {
        atomicAdd(&D[ni[i]], 1.0f);
        atomicAdd(&B[hi[i]], 1.0f);
    }
}

__global__ __launch_bounds__(256) void k_invert(float* __restrict__ p, int n) {
    int i = blockIdx.x * 256 + threadIdx.x;
    if (i < n) { float v = p[i]; p[i] = (v > 0.0f) ? 1.0f / v : 0.0f; }
}

// ---------------- WMMA GEMM: Out[M,N] = A[M,K] * W[K,N] (bf16 in, f32 acc) ----
// K = 1 << kshift (512 or 256). One 16x16 tile per wave, 8 waves/block.

__global__ __launch_bounds__(256) void k_gemm_bf16_wmma(const float* __restrict__ A,
                                                        const float* __restrict__ W,
                                                        float* __restrict__ Out,
                                                        int M, int N, int kshift) {
    __shared__ __bf16 ldsWt[16 * (IN_CH + W_PAD)];  // transposed W slab: [n][k], padded
    __shared__ __bf16 ldsA[8][16 * A_RS];           // per-wave A tile: [r][k], padded

    const int K    = 1 << kshift;
    const int RS   = K + W_PAD;                     // ldsWt row stride (halves)
    const int tid  = threadIdx.x;
    const int lane = tid & 31;
    const int wave = tid >> 5;
    const int n0   = blockIdx.x * 16;

    // cooperative load of transposed W slab, packed bf16x2 stores (one b32 each)
    {
        const int kpsh = kshift - 1;                // K/2 pairs per n-row
        const int KP   = K >> 1;
        for (int i = tid; i < 16 * KP; i += 256) {
            int n  = i >> kpsh;
            int kp = i & (KP - 1);
            int k  = kp << 1;
            bf16x2 p;
            p[0] = (__bf16)W[(size_t)k       * N + n0 + n];
            p[1] = (__bf16)W[(size_t)(k + 1) * N + n0 + n];
            ((bf16x2*)ldsWt)[n * ((RS) >> 1) + kp] = p;
        }
    }
    __syncthreads();

    const int mtile = blockIdx.y * 8 + wave;
    if (mtile * 16 >= M) return;                    // wave-uniform: EXEC stays all-1s
    const int m0  = mtile * 16;
    const int hi  = lane >> 4;                      // lane half selects K group (+8/+24)
    const int r16 = lane & 15;

    // software-pipelined K loop: global float2 loads for tile kk+32 are issued
    // before the WMMA for tile kk (single LDS buffer is safe: DS in-order/wave).
    float2 f[8];
    {   // prologue: fetch tile kk = 0
        #pragma unroll
        for (int it = 0; it < 8; ++it) {
            int i = it * 32 + lane, r = i >> 4, kp = i & 15;
            f[it] = *(const float2*)&A[(size_t)(m0 + r) * K + (kp << 1)];
        }
    }

    v8f acc = {};
    for (int kk = 0; kk < K; kk += 32) {
        // stage current tile into LDS (packed bf16x2 -> ds_store_b32)
        #pragma unroll
        for (int it = 0; it < 8; ++it) {
            int i = it * 32 + lane, r = i >> 4, kp = i & 15;
            bf16x2 p;
            p[0] = (__bf16)f[it].x;
            p[1] = (__bf16)f[it].y;
            ((bf16x2*)(&ldsA[wave][0]))[r * (A_RS >> 1) + kp] = p;
        }
        // prefetch next tile from global while WMMA pipeline drains
        int kn = kk + 32;
        float2 g[8];
        if (kn < K) {
            #pragma unroll
            for (int it = 0; it < 8; ++it) {
                int i = it * 32 + lane, r = i >> 4, kp = i & 15;
                g[it] = *(const float2*)&A[(size_t)(m0 + r) * K + kn + (kp << 1)];
            }
        }
        // fragment loads: two contiguous 16B runs each -> ds_load_b128 x2
        bf16x8 a0 = *(const bf16x8*)&ldsA[wave][r16 * A_RS + (hi << 3)];
        bf16x8 a1 = *(const bf16x8*)&ldsA[wave][r16 * A_RS + 16 + (hi << 3)];
        bf16x8 b0 = *(const bf16x8*)&ldsWt[r16 * RS + kk + (hi << 3)];
        bf16x8 b1 = *(const bf16x8*)&ldsWt[r16 * RS + kk + 16 + (hi << 3)];
        v16bf a = __builtin_shufflevector(a0, a1, 0,1,2,3,4,5,6,7,8,9,10,11,12,13,14,15);
        v16bf b = __builtin_shufflevector(b0, b1, 0,1,2,3,4,5,6,7,8,9,10,11,12,13,14,15);
        acc = __builtin_amdgcn_wmma_f32_16x16x32_bf16(
            /*neg_a=*/false, a, /*neg_b=*/false, b,
            /*c_mod=*/(short)0, acc, /*reuse_a=*/false, /*reuse_b=*/false);
        #pragma unroll
        for (int it = 0; it < 8; ++it) f[it] = g[it];
    }

    // C/D layout: VGPR v, lane: M = v + 8*hi, N = lane&15
    #pragma unroll
    for (int v = 0; v < 8; ++v) {
        int row = m0 + v + hi * 8;
        Out[(size_t)row * N + n0 + r16] = acc[v];
    }
}

// ---------------- sparse aggregation kernels ----------------

// float4-vectorized: dst[di[e]*C + c..c+3] += src[si[e]*C + c..c+3]
__global__ __launch_bounds__(256) void k_scatter_add4(const float* __restrict__ src,
                                                      const int* __restrict__ si,
                                                      const int* __restrict__ di,
                                                      float* __restrict__ dst,
                                                      long long total4, int cshift) {
    long long i = (long long)blockIdx.x * 256 + threadIdx.x;
    if (i >= total4) return;
    const int cs4 = cshift - 2;
    long long e = i >> cs4;
    int c = (int)(i & ((1 << cs4) - 1)) << 2;
    const float4 v = *(const float4*)&src[((size_t)si[e] << cshift) + c];
    float* d = &dst[((size_t)di[e] << cshift) + c];
    atomicAdd(d + 0, v.x);
    atomicAdd(d + 1, v.y);
    atomicAdd(d + 2, v.z);
    atomicAdd(d + 3, v.w);
}

__global__ __launch_bounds__(256) void k_row_scale(float* __restrict__ data,
                                                   const float* __restrict__ rs,
                                                   long long total, int cshift) {
    long long i = (long long)blockIdx.x * 256 + threadIdx.x;
    if (i >= total) return;
    data[i] *= rs[i >> cshift];
}

__global__ __launch_bounds__(256) void k_scale_bias_act(float* __restrict__ data,
                                                        const float* __restrict__ rs,
                                                        const float* __restrict__ bias,
                                                        long long total, int cshift, int relu) {
    long long i = (long long)blockIdx.x * 256 + threadIdx.x;
    if (i >= total) return;
    int c = (int)(i & ((1 << cshift) - 1));
    float v = data[i] * rs[i >> cshift] + bias[c];
    data[i] = relu ? fmaxf(v, 0.0f) : v;
}

// ---------------- log-softmax over 64 channels, one wave32 per row -----------

__global__ __launch_bounds__(256) void k_log_softmax64(const float* __restrict__ in,
                                                       float* __restrict__ out, int rows) {
    int row  = blockIdx.x * 8 + (threadIdx.x >> 5);
    int lane = threadIdx.x & 31;
    if (row >= rows) return;
    size_t base = (size_t)row * 64;
    float v0 = in[base + lane];
    float v1 = in[base + 32 + lane];
    float m = fmaxf(v0, v1);
    #pragma unroll
    for (int off = 16; off > 0; off >>= 1) m = fmaxf(m, __shfl_xor(m, off, 32));
    float s = __expf(v0 - m) + __expf(v1 - m);
    #pragma unroll
    for (int off = 16; off > 0; off >>= 1) s += __shfl_xor(s, off, 32);
    float ls = __logf(s);
    out[base + lane]      = v0 - m - ls;
    out[base + 32 + lane] = v1 - m - ls;
}

// ---------------- launcher ----------------

extern "C" void kernel_launch(void* const* d_in, const int* in_sizes, int n_in,
                              void* d_out, int out_size, void* d_ws, size_t ws_size,
                              hipStream_t stream) {
    (void)in_sizes; (void)n_in; (void)out_size; (void)ws_size;

    const float* x  = (const float*)d_in[0];
    const int*   ei = (const int*)d_in[1];      // [2, NNZ] flat
    const float* W1 = (const float*)d_in[2];
    const float* b1 = (const float*)d_in[3];
    const float* W2 = (const float*)d_in[4];
    const float* b2 = (const float*)d_in[5];
    float* out = (float*)d_out;

    const int* node_idx  = ei;
    const int* hedge_idx = ei + NNZV;

    // workspace layout (reused across layers)
    float* Dinv = (float*)d_ws;                          // N_NODES
    float* Binv = Dinv + N_NODES;                        // N_HEDGES
    float* h1   = Binv + N_HEDGES;                       // 50000*256
    float* m1   = h1 + (size_t)N_NODES * HID_CH;         // 50000*256
    float* x2   = m1 + (size_t)N_HEDGES * HID_CH;        // 50000*256
    float* h2   = h1;                                    // 50000*64 (aliases h1)
    float* m2   = m1;                                    // 50000*64 (aliases m1)
    float* agg2 = h1 + (size_t)N_NODES * OUT_CH;         // disjoint from h2 within h1 region

    auto nb = [](long long n) { return (unsigned)((n + 255) / 256); };

    const long long nDeg = N_NODES + N_HEDGES;
    const long long nH1  = (long long)N_NODES  * HID_CH;
    const long long nM1  = (long long)N_HEDGES * HID_CH;
    const long long nH2  = (long long)N_NODES  * OUT_CH;
    const long long nM2  = (long long)N_HEDGES * OUT_CH;
    const long long t1_4 = (long long)NNZV * HID_CH / 4;
    const long long t2_4 = (long long)NNZV * OUT_CH / 4;
    const unsigned  mblk = (unsigned)((N_NODES / 16 + 7) / 8);   // 391 blocks of 8 M-tiles

    // degrees: D, B then invert in place
    k_zero<<<nb(nDeg), 256, 0, stream>>>(Dinv, nDeg);
    k_degrees<<<nb(NNZV), 256, 0, stream>>>(node_idx, hedge_idx, Dinv, Binv, NNZV);
    k_invert<<<nb(nDeg), 256, 0, stream>>>(Dinv, (int)nDeg);

    // ----- layer 1 -----
    k_gemm_bf16_wmma<<<dim3(HID_CH / 16, mblk), 256, 0, stream>>>(x, W1, h1, N_NODES, HID_CH, 9);
    k_zero<<<nb(nM1), 256, 0, stream>>>(m1, nM1);
    k_scatter_add4<<<nb(t1_4), 256, 0, stream>>>(h1, node_idx, hedge_idx, m1, t1_4, 8);
    k_row_scale<<<nb(nM1), 256, 0, stream>>>(m1, Binv, nM1, 8);
    k_zero<<<nb(nH1), 256, 0, stream>>>(x2, nH1);
    k_scatter_add4<<<nb(t1_4), 256, 0, stream>>>(m1, hedge_idx, node_idx, x2, t1_4, 8);
    k_scale_bias_act<<<nb(nH1), 256, 0, stream>>>(x2, Dinv, b1, nH1, 8, 1);

    // ----- layer 2 -----
    k_gemm_bf16_wmma<<<dim3(OUT_CH / 16, mblk), 256, 0, stream>>>(x2, W2, h2, N_NODES, OUT_CH, 8);
    k_zero<<<nb(nM2), 256, 0, stream>>>(m2, nM2);
    k_scatter_add4<<<nb(t2_4), 256, 0, stream>>>(h2, node_idx, hedge_idx, m2, t2_4, 6);
    k_row_scale<<<nb(nM2), 256, 0, stream>>>(m2, Binv, nM2, 6);
    k_zero<<<nb(nH2), 256, 0, stream>>>(agg2, nH2);
    k_scatter_add4<<<nb(t2_4), 256, 0, stream>>>(m2, hedge_idx, node_idx, agg2, t2_4, 6);
    k_scale_bias_act<<<nb(nH2), 256, 0, stream>>>(agg2, Dinv, b2, nH2, 6, 0);

    // log-softmax -> d_out
    k_log_softmax64<<<(N_NODES + 7) / 8, 256, 0, stream>>>(agg2, out, N_NODES);
}
